// VectorQuantizer_48945447306060
// MI455X (gfx1250) — matrix operand order repaired and checked by reading, not compiled, not run on hardware
//
#include <hip/hip_runtime.h>
#include <hip/hip_bf16.h>

typedef __attribute__((ext_vector_type(16))) __bf16 v16bf;
typedef __attribute__((ext_vector_type(8)))  float  v8f;

#define NUM_EMB   1024
#define EMB_DIM   64
#define NROWS     32768
#define Q_ELEMS   (NROWS * EMB_DIM)        // 2097152
#define ENC_ELEMS ((size_t)NROWS * NUM_EMB) // 33554432

// ---------------------------------------------------------------------------
// Kernel 1: normalize codebook, build bf16 hi/lo splits, zero accumulators.
// ---------------------------------------------------------------------------
__global__ __launch_bounds__(256) void vq_prep(
    const float* __restrict__ emb, float* __restrict__ ne,
    __bf16* __restrict__ ne_hi, __bf16* __restrict__ ne_lo,
    float* __restrict__ emb_sq, int* __restrict__ counts,
    float* __restrict__ loss_accum)
{
  const int k = blockIdx.x * 256 + threadIdx.x;   // 0..1023
  if (k < NUM_EMB) {
    const float* e = emb + (size_t)k * EMB_DIM;
    float s = 0.0f;
#pragma unroll
    for (int d = 0; d < EMB_DIM; ++d) s += e[d] * e[d];
    const float inv = rsqrtf(s + 1e-12f);
#pragma unroll
    for (int d = 0; d < EMB_DIM; ++d) {
      float v = e[d] * inv;
      ne[(size_t)k * EMB_DIM + d] = v;
      __bf16 h = (__bf16)v;
      ne_hi[(size_t)k * EMB_DIM + d] = h;
      ne_lo[(size_t)k * EMB_DIM + d] = (__bf16)(v - (float)h);
    }
    emb_sq[k] = s * inv * inv;   // == sum(ne*ne)
    counts[k] = 0;
  }
  if (blockIdx.x == 0 && threadIdx.x == 0) *loss_accum = 0.0f;
}

// ---------------------------------------------------------------------------
// Kernel 2: distance GEMM via v_wmma_f32_16x16x32_bf16 (hi/lo split => ~fp32
// accuracy) + per-row argmin. One wave = 16 rows; 8 waves/block; 256 blocks.
// ---------------------------------------------------------------------------
__global__ __launch_bounds__(256) void vq_argmin_wmma(
    const float* __restrict__ x, const __bf16* __restrict__ ne_hi,
    const __bf16* __restrict__ ne_lo, const float* __restrict__ emb_sq,
    int* __restrict__ indices, int* __restrict__ counts)
{
  const int lane = threadIdx.x & 31;
  const int wave = threadIdx.x >> 5;
  const int row_tile = blockIdx.x * 8 + wave;
  const int m = lane & 15;
  const int abase = (lane < 16) ? 0 : 16;  // A-layout K base (contiguous, matches B)
  const int bkoff = (lane < 16) ? 0 : 16;  // B-layout K base (32x16 bf16 B)
  const int row = row_tile * 16 + m;

  // Build A fragments: hi and lo splits for K-blocks 0 (K=0..31) and 1 (32..63)
  v16bf ah0, ah1, al0, al1;
  const float* xr = x + (size_t)row * EMB_DIM;
#pragma unroll
  for (int i = 0; i < 8; ++i) {
    float v0 = xr[abase + i];
    float v1 = xr[abase + 8 + i];
    float v2 = xr[32 + abase + i];
    float v3 = xr[32 + abase + 8 + i];
    __bf16 h0 = (__bf16)v0, h1 = (__bf16)v1, h2 = (__bf16)v2, h3 = (__bf16)v3;
    ah0[i] = h0; ah0[8 + i] = h1; ah1[i] = h2; ah1[8 + i] = h3;
    al0[i] = (__bf16)(v0 - (float)h0); al0[8 + i] = (__bf16)(v1 - (float)h1);
    al1[i] = (__bf16)(v2 - (float)h2); al1[8 + i] = (__bf16)(v3 - (float)h3);
  }

  float best[8];
  int   bidx[8];
#pragma unroll
  for (int j = 0; j < 8; ++j) { best[j] = 3.4e38f; bidx[j] = 0x7fffffff; }

  for (int t = 0; t < NUM_EMB / 16; ++t) {
    const int code = t * 16 + m;                 // this lane's column code
    const __bf16* bp = ne_hi + (size_t)code * EMB_DIM + bkoff;
    const __bf16* lp = ne_lo + (size_t)code * EMB_DIM + bkoff;
    v16bf bh0 = *(const v16bf*)(bp);
    v16bf bh1 = *(const v16bf*)(bp + 32);
    v16bf bl0 = *(const v16bf*)(lp);
    v16bf bl1 = *(const v16bf*)(lp + 32);

    v8f c = {};
    // dot = xh*eh + xh*el + xl*eh  (drop xl*el; ~fp32 accuracy)
    c = __builtin_amdgcn_wmma_f32_16x16x32_bf16(false, ah0, false, bh0, (short)0, c, false, false);
    c = __builtin_amdgcn_wmma_f32_16x16x32_bf16(false, ah0, false, bl0, (short)0, c, false, false);
    c = __builtin_amdgcn_wmma_f32_16x16x32_bf16(false, al0, false, bh0, (short)0, c, false, false);
    c = __builtin_amdgcn_wmma_f32_16x16x32_bf16(false, ah1, false, bh1, (short)0, c, false, false);
    c = __builtin_amdgcn_wmma_f32_16x16x32_bf16(false, ah1, false, bl1, (short)0, c, false, false);
    c = __builtin_amdgcn_wmma_f32_16x16x32_bf16(false, al1, false, bh1, (short)0, c, false, false);

    const float esq = emb_sq[code];
#pragma unroll
    for (int j = 0; j < 8; ++j) {             // row M = j + 8*(lane>=16)
      float s = esq - 2.0f * c[j];
      if (s < best[j]) { best[j] = s; bidx[j] = code; }  // ascending codes => first-min kept
    }
  }

  // argmin across the 16 columns (lanes differing in bits 0..3), index tie-break
#pragma unroll
  for (int mask = 1; mask < 16; mask <<= 1) {
#pragma unroll
    for (int j = 0; j < 8; ++j) {
      float os = __shfl_xor(best[j], mask, 32);
      int   oi = __shfl_xor(bidx[j], mask, 32);
      if (os < best[j] || (os == best[j] && oi < bidx[j])) { best[j] = os; bidx[j] = oi; }
    }
  }

  if (m == 0) {   // lane 0 -> rows 0..7 of tile, lane 16 -> rows 8..15
    const int rbase = row_tile * 16 + ((lane < 16) ? 0 : 8);
#pragma unroll
    for (int j = 0; j < 8; ++j) {
      indices[rbase + j] = bidx[j];
      atomicAdd(&counts[bidx[j]], 1);
    }
  }
}

// ---------------------------------------------------------------------------
// Kernel 3: stream outputs. One block per row. One-hot (134 MB) uses
// non-temporal stores so it doesn't evict codebook/x from cache.
// ---------------------------------------------------------------------------
__global__ __launch_bounds__(256) void vq_outputs(
    const float* __restrict__ x, const float* __restrict__ ne,
    const int* __restrict__ indices, float* __restrict__ out_q,
    float* __restrict__ out_enc, float* __restrict__ out_idx,
    float* __restrict__ loss_accum)
{
  const int row = blockIdx.x;
  const int tid = threadIdx.x;
  const int idx = indices[row];

  float* enc = out_enc + (size_t)row * NUM_EMB;
#pragma unroll
  for (int i = 0; i < 4; ++i) {
    int col = tid + i * 256;
    __builtin_nontemporal_store(col == idx ? 1.0f : 0.0f, enc + col);
  }

  if (tid < EMB_DIM) {
    float q = ne[(size_t)idx * EMB_DIM + tid];
    __builtin_nontemporal_store(q, out_q + (size_t)row * EMB_DIM + tid);
    float d = q - x[(size_t)row * EMB_DIM + tid];
    float p = d * d;
#pragma unroll
    for (int mask = 16; mask >= 1; mask >>= 1) p += __shfl_xor(p, mask, 32);
    if ((tid & 31) == 0) atomicAdd(loss_accum, p);
  }
  if (tid == 0) out_idx[row] = (float)idx;
}

// ---------------------------------------------------------------------------
// Kernel 4: loss scalar + unique code count.
// ---------------------------------------------------------------------------
__global__ __launch_bounds__(256) void vq_finalize(
    const int* __restrict__ counts, const float* __restrict__ loss_accum,
    const float* __restrict__ cost, float* __restrict__ out_loss,
    float* __restrict__ out_unique)
{
  __shared__ int s_unique;
  if (threadIdx.x == 0) s_unique = 0;
  __syncthreads();
  int cnt = 0;
  for (int k = threadIdx.x; k < NUM_EMB; k += 256) cnt += (counts[k] > 0) ? 1 : 0;
  atomicAdd(&s_unique, cnt);
  __syncthreads();
  if (threadIdx.x == 0) {
    *out_unique = (float)s_unique;
    // loss = q_loss + cost*e_loss = (1 + cost) * mse
    *out_loss = (1.0f + cost[0]) * (*loss_accum) / (float)Q_ELEMS;
  }
}

// ---------------------------------------------------------------------------
extern "C" void kernel_launch(void* const* d_in, const int* in_sizes, int n_in,
                              void* d_out, int out_size, void* d_ws, size_t ws_size,
                              hipStream_t stream) {
  const float* x    = (const float*)d_in[0];   // 32*32*32*64
  const float* emb  = (const float*)d_in[1];   // 1024*64
  const float* cost = (const float*)d_in[2];   // 1

  char* ws = (char*)d_ws;
  float*  ne         = (float*)(ws);                     // 262144 B
  __bf16* ne_hi      = (__bf16*)(ws + 262144);           // 131072 B
  __bf16* ne_lo      = (__bf16*)(ws + 393216);           // 131072 B
  float*  emb_sq     = (float*)(ws + 524288);            //   4096 B
  int*    counts     = (int*)(ws + 528384);              //   4096 B
  float*  loss_accum = (float*)(ws + 532480);            //    128 B pad
  int*    indices    = (int*)(ws + 532608);              // 131072 B

  float* out      = (float*)d_out;
  float* out_q    = out;                                  // [0, 2097152)
  float* out_loss = out + Q_ELEMS;                        // [2097152]
  float* out_uni  = out + Q_ELEMS + 1;                    // [2097153]
  float* out_enc  = out + Q_ELEMS + 2;                    // [2097154, +33554432)
  float* out_idx  = out + Q_ELEMS + 2 + ENC_ELEMS;        // [35651586, +32768)

  vq_prep<<<4, 256, 0, stream>>>(emb, ne, ne_hi, ne_lo, emb_sq, counts, loss_accum);
  vq_argmin_wmma<<<256, 256, 0, stream>>>(x, ne_hi, ne_lo, emb_sq, indices, counts);
  vq_outputs<<<NROWS, 256, 0, stream>>>(x, ne, indices, out_q, out_enc, out_idx, loss_accum);
  vq_finalize<<<1, 256, 0, stream>>>(counts, loss_accum, cost, out_loss, out_uni);
}
